// Tokenizer_45011257262125
// MI455X (gfx1250) — matrix operand order repaired and checked by reading, not compiled, not run on hardware
//
#include <hip/hip_runtime.h>
#include <cstdint>

// CDNA5 WMMA / TDM vector types
typedef __attribute__((ext_vector_type(16))) __bf16 v16bf;
typedef __attribute__((ext_vector_type(8)))  float  v8f;
typedef __attribute__((ext_vector_type(4)))  unsigned v4u;
typedef __attribute__((ext_vector_type(8)))  int      v8i;
typedef __attribute__((ext_vector_type(4)))  int      v4i;

#define BATCH     32
#define NUM_VARS  16
#define SAMPLES   4096
#define EMBED     128
#define WIN       32
#define STEP      4
#define NTOK      1016          // (SAMPLES - WIN) / STEP
#define KDIM      (WIN * EMBED) // 4096
#define NSTAGES   32            // 128 K-elems (4 WMMA K-steps) per stage
#define ROWPAD    136           // 128 elems + 8 elems (16B) LDS pad per row

// -------- workspace layout --------
// [0, 32MB)    : enc  bf16 [B][S][D]
// [32MB, 33MB) : Bpack u32 [kT][nT][lane][j]
#define ENC_BYTES ((size_t)BATCH * SAMPLES * EMBED * 2)

// ---------------------------------------------------------------------------
// Kernel 1: enc[b,s,d] = floor(sum_v x[b,v,s] * W_sample[v,d] + b_sample[d])
// stored as bf16 (exact: values are small integers).
// ---------------------------------------------------------------------------
__global__ __launch_bounds__(128) void enc_kernel(const float* __restrict__ x,
                                                  const float* __restrict__ Ws,
                                                  const float* __restrict__ bs,
                                                  uint16_t* __restrict__ enc) {
  const int S_CHUNK = 64;
  const int blk = blockIdx.x;            // BATCH * (SAMPLES/64) blocks
  const int b   = blk >> 6;
  const int s0  = (blk & 63) * S_CHUNK;
  const int d   = threadIdx.x;           // 0..127

  __shared__ float xs[NUM_VARS][S_CHUNK];
  for (int i = threadIdx.x; i < NUM_VARS * S_CHUNK; i += 128) {
    const int v = i >> 6, s = i & 63;
    xs[v][s] = x[((size_t)b * NUM_VARS + v) * SAMPLES + s0 + s];
  }
  __syncthreads();

  float w[NUM_VARS];
#pragma unroll
  for (int v = 0; v < NUM_VARS; ++v) w[v] = Ws[v * EMBED + d];
  const float bsd = bs[d];

  for (int s = 0; s < S_CHUNK; ++s) {
    float acc = bsd;
#pragma unroll
    for (int v = 0; v < NUM_VARS; ++v) acc = fmaf(xs[v][s], w[v], acc);
    enc[((size_t)b * SAMPLES + s0 + s) * EMBED + d] =
        __builtin_bit_cast(uint16_t, (__bf16)floorf(acc));
  }
}

// ---------------------------------------------------------------------------
// Kernel 2: pre-swizzle W_patch into the CDNA5 bf16 B-fragment layout
// (B tile 32Kx16N; VGPR j lanes 0-15: K=(j>>2)*16+2(j&3),+1; lanes 16-31: K=(j>>2)*16+8+2(j&3),+1),
// stored so each lane reads 8 contiguous dwords.
// ---------------------------------------------------------------------------
__global__ __launch_bounds__(256) void packB_kernel(const float* __restrict__ Wp,
                                                    uint32_t* __restrict__ Bp) {
  const int tid  = blockIdx.x * 256 + threadIdx.x;  // 0 .. 262143
  const int j    = tid & 7;
  const int lane = (tid >> 3) & 31;
  const int nT   = (tid >> 8) & 7;
  const int kT   = tid >> 11;
  const int n = nT * 16 + (lane & 15);
  const int k = kT * 32 + (j >> 2) * 16 + (lane >> 4) * 8 + 2 * (j & 3);
  const uint16_t lo = __builtin_bit_cast(uint16_t, (__bf16)Wp[(size_t)k * EMBED + n]);
  const uint16_t hi = __builtin_bit_cast(uint16_t, (__bf16)Wp[(size_t)(k + 1) * EMBED + n]);
  Bp[tid] = ((uint32_t)hi << 16) | (uint32_t)lo;
}

// ---------------------------------------------------------------------------
// TDM issue: DMA a 16-row x 128-elem bf16 tile (row stride 512 elems) from
// global to LDS, padding the LDS destination by 4 dwords every 64 dwords
// (row stride 272B -> bank-conflict-free ds_load_b128 fragment reads).
// 6-arg builtin form (clang-23 toolchain): (g0, g1, g2, g3, g4, cpol).
// ---------------------------------------------------------------------------
__device__ __forceinline__ void tdm_load_tile(const uint16_t* gsrc, unsigned lds_off) {
  const uint64_t ga = (uint64_t)(uintptr_t)gsrc;
  v4u g0;
  g0.x = 1u;                                               // count=1 (user D#)
  g0.y = lds_off;                                          // LDS byte address
  g0.z = (unsigned)(ga & 0xFFFFFFFFu);                     // global_addr[31:0]
  g0.w = (unsigned)((ga >> 32) & 0x01FFFFFFu) | (2u << 30);// global_addr[56:32] | type=2
  v8i g1;
  g1[0] = (int)0x07510000u;   // data_size=2B | pad_enable | pad_interval=64dw | pad_amount=4dw
  g1[1] = (int)(128u << 16);  // tensor_dim0 = 128
  g1[2] = (int)(16u << 16);   // tensor_dim1 = 16
  g1[3] = (int)(128u << 16);  // tile_dim0 = 128
  g1[4] = 16;                 // tile_dim1 = 16, tile_dim2 = 0 (2D)
  g1[5] = 512;                // tensor_dim0_stride = 512 elems (token row stride)
  g1[6] = 0;
  g1[7] = 0;
  const v4i z4 = {0, 0, 0, 0};
  const v8i z8 = {0, 0, 0, 0, 0, 0, 0, 0};
  __builtin_amdgcn_tensor_load_to_lds(g0, g1, z4, z4, z8, 0);
}

// ---------------------------------------------------------------------------
// Kernel 3: tokens[b,t,n] = floor( windows[b,t,:] @ W_patch[:,n] + b_patch[n] )
// One block = 16 token rows x 128 cols; 8 waves = 8 N-tiles. A tile is DMA'd
// once per stage by wave 0 via TDM (double-buffered LDS), consumed by all 8
// waves as WMMA A-fragments via ds_load_b128; B comes pre-swizzled from L2.
// ---------------------------------------------------------------------------
__global__ __launch_bounds__(256) void gemm_kernel(const uint16_t* __restrict__ enc,
                                                   const uint32_t* __restrict__ Bp,
                                                   const float* __restrict__ bpatch,
                                                   float* __restrict__ out) {
  const int lane = threadIdx.x & 31;
  const int wave = threadIdx.x >> 5;   // 0..7 = N tile
  const int b    = blockIdx.x >> 6;
  const int mt   = blockIdx.x & 63;    // M tile (16 token rows) within batch
  const int half = lane >> 4;
  const int m    = lane & 15;

  __shared__ __align__(16) uint16_t abuf[2][16 * ROWPAD];  // 2 x 4.25KB

  const uint16_t* __restrict__ encb =
      enc + (size_t)b * SAMPLES * EMBED + (size_t)(mt * 16) * (STEP * EMBED);
  const uint32_t* __restrict__ bq = Bp + wave * 256 + lane * 8;

  union ABits { uint4 u[2]; v16bf v; };
  v8f acc = {0.f, 0.f, 0.f, 0.f, 0.f, 0.f, 0.f, 0.f};

  if (wave == 0)
    tdm_load_tile(encb, (unsigned)(uintptr_t)&abuf[0][0]);

  for (int st = 0; st < NSTAGES; ++st) {
    __syncthreads();                       // prior readers of buf[(st+1)&1] done
    if (wave == 0) {
      if (st + 1 < NSTAGES) {
        tdm_load_tile(encb + (st + 1) * 128,
                      (unsigned)(uintptr_t)&abuf[(st + 1) & 1][0]);
        __builtin_amdgcn_s_wait_tensorcnt(1);  // stage st complete (in-order TDM)
      } else {
        __builtin_amdgcn_s_wait_tensorcnt(0);
      }
    }
    __syncthreads();                       // buf[st&1] visible to all waves

    const uint16_t* ab = &abuf[st & 1][0];
    __builtin_prefetch(bq + 4 * 2048, 0, 0);   // B pack, one stage ahead
#pragma unroll
    for (int ku = 0; ku < 4; ++ku) {
      ABits a, bb;
      // A-fragment (16-bit A 16x32): lane(M=m, half) holds K = half*8+[0..7]
      // and K = 16+half*8+[0..7] -> two contiguous 16B LDS chunks.
      a.u[0]  = *(const uint4*)(ab + m * ROWPAD + ku * 32 + half * 8);
      a.u[1]  = *(const uint4*)(ab + m * ROWPAD + ku * 32 + 16 + half * 8);
      bb.u[0] = *(const uint4*)(bq);
      bb.u[1] = *(const uint4*)(bq + 4);
      acc = __builtin_amdgcn_wmma_f32_16x16x32_bf16(
          /*neg_a=*/false, a.v, /*neg_b=*/false, bb.v,
          /*c_mod=*/(short)0, acc, /*reuse_a=*/false, /*reuse_b=*/false);
      bq += 8 * 32 * 8;  // next K tile of the pack
    }
  }

  // C/D layout: VGPR j -> row M = j + half*8 ; lane%16 -> column N
  const int n = wave * 16 + (lane & 15);
  const float bpn = bpatch[n];
#pragma unroll
  for (int j = 0; j < 8; ++j) {
    const int t = mt * 16 + j + half * 8;
    if (t < NTOK) {
      out[((size_t)b * NTOK + t) * EMBED + n] = floorf(acc[j] + bpn);
    }
  }
}

extern "C" void kernel_launch(void* const* d_in, const int* in_sizes, int n_in,
                              void* d_out, int out_size, void* d_ws, size_t ws_size,
                              hipStream_t stream) {
  const float* x  = (const float*)d_in[0];  // [32,16,4096]
  const float* Ws = (const float*)d_in[1];  // [16,128]
  const float* bs = (const float*)d_in[2];  // [128]
  const float* Wp = (const float*)d_in[3];  // [4096,128]
  const float* bp = (const float*)d_in[4];  // [128]
  float* out = (float*)d_out;               // [32,1016,128]

  uint16_t* enc = (uint16_t*)d_ws;
  uint32_t* Bp  = (uint32_t*)((char*)d_ws + ENC_BYTES);

  enc_kernel<<<BATCH * (SAMPLES / 64), 128, 0, stream>>>(x, Ws, bs, enc);
  packB_kernel<<<(KDIM * EMBED / 2) / 256, 256, 0, stream>>>(Wp, Bp);
  gemm_kernel<<<BATCH * 64, 256, 0, stream>>>(enc, Bp, bp, out);
}
